// AxialAttention_60172491817368
// MI455X (gfx1250) — matrix operand order repaired and checked by reading, not compiled, not run on hardware
//
#include <hip/hip_runtime.h>
#include <hip/hip_bf16.h>
#include <math.h>
#include <stdint.h>

// ---------------------------------------------------------------------------
// Axial attention for MI455X (gfx1250).
// GEMMs: v_wmma_f32_16x16x32_bf16.  Tile staging: Tensor Data Mover
// (tensor_load_to_lds + s_wait_tensorcnt), double-buffered in LDS.
// Sizes fixed per reference: B=8, C=256, H=W=128, HEADS=8, DHEAD=64, INNER=512.
// ---------------------------------------------------------------------------

typedef __bf16 bf16;
typedef __attribute__((ext_vector_type(16))) __bf16 v16bf;
typedef __attribute__((ext_vector_type(8)))  float  v8f;
typedef __attribute__((ext_vector_type(8)))  __bf16 bf16x8;
typedef __attribute__((ext_vector_type(4)))  float  f32x4;
typedef __attribute__((ext_vector_type(4)))  unsigned int u32x4;
typedef __attribute__((ext_vector_type(4)))  int    i32x4v;
typedef __attribute__((ext_vector_type(8)))  int    i32x8v;

#define NHEADS 8
#define DHEAD  64
#define INNER  512
#define CIN    256
#define BATCH  8
#define HDIM   128
#define WDIM   128
#define NPIX   (HDIM * WDIM)     // 16384 pixels per batch image
#define FEAT   (WDIM * DHEAD)    // 8192 attention feature dim
#define BH_STRIDE ((size_t)HDIM * FEAT)  // elems per (b,head) q/k/v block

#if defined(__has_include)
#  if __has_include(<hip/amd_detail/amd_gfx1250_TDM.h>)
#    define TDM_SIX_ARG 1
#  endif
#endif
#ifndef TDM_SIX_ARG
#  define TDM_SIX_ARG 0
#endif

// D = A(16x32 bf16) * B(32x16 bf16) + C(16x16 f32)
__device__ __forceinline__ v8f wmma_bf16(v16bf a, v16bf b, v8f c) {
  return __builtin_amdgcn_wmma_f32_16x16x32_bf16(
      false, a, false, b, (short)0, c, false, false);
}

// ---------------------------------------------------------------------------
// Tensor Data Mover: load a strided tile (up to 3D) from global into LDS.
// Descriptor bit layout per CDNA5 ISA §8 (D# groups 0..3).
//   dsz_code: 0=1B 1=2B 2=4B.  tile_d2==0 -> 2D tile.
//   tensor_dim0/1/2 are set large (1<<20) so no OOB zero-fill triggers.
// ---------------------------------------------------------------------------
__device__ __forceinline__ void tdm_load_tile(
    uint32_t lds_addr, const void* gptr,
    uint32_t tile_d0, uint32_t tile_d1, uint32_t tile_d2,
    uint64_t stride0, uint64_t stride1, uint32_t dsz_code) {
  const uint64_t ga = (uint64_t)(uintptr_t)gptr;
  u32x4 g0;
  g0[0] = 1u;                                   // count=1 (valid user D#)
  g0[1] = lds_addr;                             // LDS byte address
  g0[2] = (uint32_t)ga;                         // global_addr[31:0]
  g0[3] = (uint32_t)((ga >> 32) & 0x01FFFFFFu)  // global_addr[56:32]
          | (2u << 30);                         // type=2 ("image")
  i32x8v g1;
  g1[0] = (int)(dsz_code << 16);                // wg_mask=0 | data_size
  g1[1] = 0;                                    // tensor_dim0 lo16 (dim0 = 1<<20)
  g1[2] = 16;                                   // tensor_dim0 hi16 | tensor_dim1 lo16
  g1[3] = (int)((tile_d0 << 16) | 16u);         // tensor_dim1 hi16 | tile_dim0
  g1[4] = (int)(tile_d1 | (tile_d2 << 16));     // tile_dim1 | tile_dim2
  g1[5] = (int)(uint32_t)stride0;               // tensor_dim0_stride[31:0]
  g1[6] = (int)((uint32_t)((stride0 >> 32) & 0xFFFFu) |
                (((uint32_t)stride1 & 0xFFFFu) << 16)); // s0[47:32] | s1[15:0]
  g1[7] = (int)(uint32_t)(stride1 >> 16);       // tensor_dim1_stride[47:16]
  i32x4v g2;
  g2[0] = (int)(1u << 20);                      // tensor_dim2 (large)
  g2[1] = 1;                                    // tensor_dim3
  g2[2] = 0;                                    // tensor_dim2_stride lo32
  g2[3] = 0;                                    // s2 hi | tile_dim3=0
  i32x4v g3;
  g3[0] = 0;                                    // tensor_dim3_stride lo32
  g3[1] = (int)(1u << 16);                      // s3 hi16 | tensor_dim4=1 (lo16)
  g3[2] = 0;                                    // tensor_dim4 hi16 | tile_dim4=0
  g3[3] = 0;
#if TDM_SIX_ARG
  i32x8v g4 = {};
  __builtin_amdgcn_tensor_load_to_lds(g0, g1, g2, g3, g4, 0);
#else
  __builtin_amdgcn_tensor_load_to_lds(g0, g1, g2, g3, 0);
#endif
}

__device__ __forceinline__ uint32_t lds_off(const void* p) {
  // Generic pointers to LDS carry the byte offset in addr[31:0] (ISA §10.2).
  return (uint32_t)(uintptr_t)p;
}

// Element offset within one (128 tokens x 8192 feats) (b,head) block.
// T==0 (row attn): token=x, f=y*64+d -> token*8192+f
// T==1 (col attn): token=y, f=x*64+d -> x*8192 + y*64 + d
__device__ __forceinline__ size_t feat_off(int token, int f, int T) {
  return T ? ((size_t)(f >> 6)) * FEAT + (size_t)token * DHEAD + (f & 63)
           : (size_t)token * FEAT + (size_t)f;
}

// A-fragment (16x32 bf16): lane (g=lane>>4, lm=lane&15) holds row M=lm;
// elements j=0..15 -> K=k0+g*16+j (two 16B chunks).
__device__ __forceinline__ v16bf a_frag_from_chunks(bf16x8 c0, bf16x8 c1) {
  v16bf a;
#pragma unroll
  for (int j = 0; j < 8; ++j) { a[j] = c0[j]; a[j + 8] = c1[j]; }
  return a;
}

__device__ __forceinline__ v16bf a_frag_bf16(const bf16* base, int token, int k0,
                                             int g, int T) {
  bf16x8 c0 = *(const bf16x8*)(base + feat_off(token, k0 + g * 16, T));
  bf16x8 c1 = *(const bf16x8*)(base + feat_off(token, k0 + g * 16 + 8, T));
  return a_frag_from_chunks(c0, c1);
}

// A-fragment from an fp32 row-major weight matrix, converted to bf16.
__device__ __forceinline__ v16bf a_frag_f32(const float* row, int k0, int g) {
  f32x4 x0 = *(const f32x4*)(row + k0 + g * 16);
  f32x4 x1 = *(const f32x4*)(row + k0 + g * 16 + 4);
  f32x4 y0 = *(const f32x4*)(row + k0 + g * 16 + 8);
  f32x4 y1 = *(const f32x4*)(row + k0 + g * 16 + 12);
  v16bf a;
#pragma unroll
  for (int j = 0; j < 4; ++j) {
    a[j] = (bf16)x0[j]; a[j + 4] = (bf16)x1[j];
    a[j + 8] = (bf16)y0[j]; a[j + 12] = (bf16)y1[j];
  }
  return a;
}

// B-fragment (32x16 bf16): lane holds column N=lm, elements j=0..15 -> K=k0+g*16+j.
__device__ __forceinline__ v16bf b_frag_contig(const bf16* p) {
  bf16x8 c0 = *(const bf16x8*)p;
  bf16x8 c1 = *(const bf16x8*)(p + 8);
  v16bf b;
#pragma unroll
  for (int j = 0; j < 8; ++j) { b[j] = c0[j]; b[j + 8] = c1[j]; }
  return b;
}

// ---------------------------------------------------------------------------
// Kernel 1: qkv = w_qkv @ x.  M=1536, K=256, N=16384 per batch.
// TDM stages the shared x tile (32 ch x 64 px, f32) into a double buffer;
// all 8 waves read B-fragments from LDS instead of 8x-redundant global loads.
// ---------------------------------------------------------------------------
__global__ __launch_bounds__(256) void qkv_proj_kernel(
    const float* __restrict__ x, const float* __restrict__ wqkv,
    bf16* __restrict__ q, bf16* __restrict__ k, bf16* __restrict__ v) {
  __shared__ float Xtile[2][32 * 64];  // 2 x 8 KB
  const int lane = threadIdx.x & 31, wave = threadIdx.x >> 5;
  const int g = lane >> 4, lm = lane & 15;
  const int b  = blockIdx.z;
  const int M0 = (blockIdx.y * 8 + wave) * 16;
  const int N0 = blockIdx.x * 64;

  const float* xg = x + (size_t)b * CIN * NPIX + N0;
  const float* arow = wqkv + (size_t)(M0 + lm) * CIN;

  if (wave == 0) {
    tdm_load_tile(lds_off(&Xtile[0][0]), xg, 64, 32, 0, NPIX, 0, 2);
    __builtin_amdgcn_s_wait_tensorcnt(0);
  }
  __syncthreads();

  v8f acc[4] = {};
  for (int k0 = 0; k0 < CIN; k0 += 32) {
    const int buf = (k0 >> 5) & 1;
    if (wave == 0 && k0 + 32 < CIN)
      tdm_load_tile(lds_off(&Xtile[buf ^ 1][0]), xg + (size_t)(k0 + 32) * NPIX,
                    64, 32, 0, NPIX, 0, 2);
    v16bf a = a_frag_f32(arow, k0, g);
#pragma unroll
    for (int nt = 0; nt < 4; ++nt) {
      const float* xl = &Xtile[buf][g * 16 * 64 + nt * 16 + lm];
      v16bf bb;
#pragma unroll
      for (int j = 0; j < 16; ++j) bb[j] = (bf16)xl[j * 64];
      acc[nt] = wmma_bf16(a, bb, acc[nt]);
    }
    if (wave == 0) __builtin_amdgcn_s_wait_tensorcnt(0);
    __syncthreads();
  }

#pragma unroll
  for (int nt = 0; nt < 4; ++nt) {
    const int p  = N0 + nt * 16 + lm;
    const int xr = p >> 7, yc = p & 127;
#pragma unroll
    for (int r = 0; r < 8; ++r) {
      const int o = M0 + r + 8 * g;
      const int which = o >> 9, oc = o & 511;
      const int head = oc >> 6, d = oc & 63;
      const size_t dst =
          ((((size_t)b * NHEADS + head) * HDIM + xr) * WDIM + yc) * DHEAD + d;
      bf16* dstp = (which == 0) ? q : (which == 1) ? k : v;
      dstp[dst] = (bf16)acc[nt][r];
    }
  }
}

// ---------------------------------------------------------------------------
// Kernel 2: S = scale * Q Kᵀ (128x128 per (b,head)), softmax rows, P -> bf16.
// TDM stages the K tile (128 tokens x 32 feats) shared by all 8 waves.
// ---------------------------------------------------------------------------
__global__ __launch_bounds__(256) void attn_scores_kernel(
    const bf16* __restrict__ q, const bf16* __restrict__ k,
    bf16* __restrict__ P, int T) {
  __shared__ float S[HDIM * HDIM];      // 64 KB
  __shared__ bf16 Ktile[2][HDIM * 32];  // 2 x 8 KB
  const int lane = threadIdx.x & 31, wave = threadIdx.x >> 5;
  const int g = lane >> 4, lm = lane & 15;
  const size_t bh = blockIdx.x;
  const bf16* qb = q + bh * BH_STRIDE;
  const bf16* kb = k + bh * BH_STRIDE;
  const uint64_t kstride = T ? (uint64_t)DHEAD : (uint64_t)FEAT;

  if (wave == 0) {
    tdm_load_tile(lds_off(&Ktile[0][0]), kb + feat_off(0, 0, T),
                  32, HDIM, 0, kstride, 0, 1);
    __builtin_amdgcn_s_wait_tensorcnt(0);
  }
  __syncthreads();

  v8f acc[8] = {};
  const int m = wave * 16 + lm;  // A-fragment token row

  for (int k0 = 0; k0 < FEAT; k0 += 32) {
    const int buf = (k0 >> 5) & 1;
    if (wave == 0 && k0 + 32 < FEAT)
      tdm_load_tile(lds_off(&Ktile[buf ^ 1][0]), kb + feat_off(0, k0 + 32, T),
                    32, HDIM, 0, kstride, 0, 1);
    v16bf a = a_frag_bf16(qb, m, k0, g, T);
#pragma unroll
    for (int nt = 0; nt < 8; ++nt) {
      const int n = nt * 16 + lm;  // B-fragment token column
      v16bf bb = b_frag_contig(&Ktile[buf][n * 32 + g * 16]);
      acc[nt] = wmma_bf16(a, bb, acc[nt]);
    }
    if (wave == 0) __builtin_amdgcn_s_wait_tensorcnt(0);
    __syncthreads();
  }

  const float scale = 0.125f;  // DIM_HEAD ** -0.5
#pragma unroll
  for (int nt = 0; nt < 8; ++nt)
#pragma unroll
    for (int r = 0; r < 8; ++r)
      S[(wave * 16 + r + 8 * g) * HDIM + nt * 16 + lm] = acc[nt][r] * scale;

  __syncthreads();

  if (threadIdx.x < HDIM) {
    float* row = S + threadIdx.x * HDIM;
    float mx = row[0];
    for (int i = 1; i < HDIM; ++i) mx = fmaxf(mx, row[i]);
    float s = 0.f;
    for (int i = 0; i < HDIM; ++i) { float e = __expf(row[i] - mx); row[i] = e; s += e; }
    const float inv = 1.f / s;
    bf16* prow = P + bh * (HDIM * HDIM) + (size_t)threadIdx.x * HDIM;
    for (int i = 0; i < HDIM; ++i) prow[i] = (bf16)(row[i] * inv);
  }
}

// ---------------------------------------------------------------------------
// Kernel 3: O = P @ V (128 tokens x 8192 feats per (b,head)).
// TDM stages P (32 KB, once) and each V k-chunk (32 tokens x 128 wg-feats;
// 2D tile in row pass, 3D tile in col pass).  Col pass fuses the
// 0.5*(oh + ow) combine in place.
// ---------------------------------------------------------------------------
__global__ __launch_bounds__(256) void attn_pv_kernel(
    const bf16* __restrict__ P, const bf16* __restrict__ v,
    bf16* __restrict__ oh, int T) {
  __shared__ bf16 Plds[HDIM * HDIM];    // 32 KB
  __shared__ bf16 Vtile[2][32 * 128];   // 2 x 8 KB
  const int lane = threadIdx.x & 31, wave = threadIdx.x >> 5;
  const int g = lane >> 4, lm = lane & 15;
  const size_t bh = blockIdx.y;
  const int fwg = blockIdx.x * 128;       // workgroup feat base
  const int f0  = fwg + wave * 16;        // wave feat strip
  const bf16* vb = v + bh * BH_STRIDE;
  const bf16* Pb = P + bh * (HDIM * HDIM);

  if (wave == 0) {
    tdm_load_tile(lds_off(&Plds[0]), Pb, HDIM * HDIM, 1, 0, HDIM * HDIM, 0, 1);
    if (!T)
      tdm_load_tile(lds_off(&Vtile[0][0]), vb + fwg, 128, 32, 0, FEAT, 0, 1);
    else
      tdm_load_tile(lds_off(&Vtile[0][0]), vb + (size_t)(fwg >> 6) * FEAT,
                    64, 32, 2, DHEAD, FEAT, 1);
    __builtin_amdgcn_s_wait_tensorcnt(0);
  }
  __syncthreads();

  v8f acc[8] = {};
  const int f  = f0 + lm;
  const int fl = f - fwg;  // local feat 0..127

  for (int kt = 0; kt < HDIM; kt += 32) {
    const int buf = (kt >> 5) & 1;
    if (wave == 0 && kt + 32 < HDIM) {
      if (!T)
        tdm_load_tile(lds_off(&Vtile[buf ^ 1][0]),
                      vb + (size_t)(kt + 32) * FEAT + fwg, 128, 32, 0, FEAT, 0, 1);
      else
        tdm_load_tile(lds_off(&Vtile[buf ^ 1][0]),
                      vb + (size_t)(fwg >> 6) * FEAT + (size_t)(kt + 32) * DHEAD,
                      64, 32, 2, DHEAD, FEAT, 1);
    }
    // B fragment: V column f, token rows kt + g*16 + j (from LDS).
    v16bf bb;
    if (!T) {
#pragma unroll
      for (int j = 0; j < 16; ++j) bb[j] = Vtile[buf][(g * 16 + j) * 128 + fl];
    } else {
#pragma unroll
      for (int j = 0; j < 16; ++j)
        bb[j] = Vtile[buf][(fl >> 6) * 2048 + (g * 16 + j) * 64 + (fl & 63)];
    }
#pragma unroll
    for (int Mt = 0; Mt < 8; ++Mt) {
      const bf16* prow = Plds + (size_t)(Mt * 16 + lm) * HDIM;
      bf16x8 c0 = *(const bf16x8*)(prow + kt + g * 16);
      bf16x8 c1 = *(const bf16x8*)(prow + kt + g * 16 + 8);
      acc[Mt] = wmma_bf16(a_frag_from_chunks(c0, c1), bb, acc[Mt]);
    }
    if (wave == 0) __builtin_amdgcn_s_wait_tensorcnt(0);
    __syncthreads();
  }

#pragma unroll
  for (int Mt = 0; Mt < 8; ++Mt)
#pragma unroll
    for (int r = 0; r < 8; ++r) {
      const int m = Mt * 16 + r + 8 * g;
      const size_t off = bh * BH_STRIDE + feat_off(m, f, T);
      if (!T) {
        oh[off] = (bf16)acc[Mt][r];
      } else {
        const float prev = (float)oh[off];
        oh[off] = (bf16)(0.5f * (prev + acc[Mt][r]));
      }
    }
}

// ---------------------------------------------------------------------------
// Kernel 4: out = w_out @ combined + b_out + x.  M=256, K=512, f32 out.
// ---------------------------------------------------------------------------
__global__ __launch_bounds__(256) void out_proj_kernel(
    const bf16* __restrict__ comb, const float* __restrict__ wout,
    const float* __restrict__ bout, const float* __restrict__ x,
    float* __restrict__ out) {
  const int lane = threadIdx.x & 31, wave = threadIdx.x >> 5;
  const int g = lane >> 4, lm = lane & 15;
  const int b  = blockIdx.z;
  const int M0 = (blockIdx.y * 8 + wave) * 16;
  const int N0 = blockIdx.x * 64;

  v8f acc[4] = {};
  const float* arow = wout + (size_t)(M0 + lm) * INNER;

  for (int k0 = 0; k0 < INNER; k0 += 32) {
    v16bf a = a_frag_f32(arow, k0, g);
#pragma unroll
    for (int nt = 0; nt < 4; ++nt) {
      const int p  = N0 + nt * 16 + lm;
      const int xr = p >> 7, yc = p & 127;
      const int c0 = k0 + g * 16;
      const int head = c0 >> 6, d0 = c0 & 63;
      const bf16* cp = comb +
          (((size_t)b * NHEADS + head) * HDIM + xr) * (size_t)(WDIM * DHEAD) +
          (size_t)yc * DHEAD + d0;
      acc[nt] = wmma_bf16(a, b_frag_contig(cp), acc[nt]);
    }
  }

#pragma unroll
  for (int nt = 0; nt < 4; ++nt) {
    const int p = N0 + nt * 16 + lm;
#pragma unroll
    for (int r = 0; r < 8; ++r) {
      const int o = M0 + r + 8 * g;
      const size_t idx = ((size_t)b * CIN + o) * NPIX + p;
      out[idx] = acc[nt][r] + bout[o] + x[idx];
    }
  }
}

// ---------------------------------------------------------------------------
// Host launch
// ---------------------------------------------------------------------------
extern "C" void kernel_launch(void* const* d_in, const int* in_sizes, int n_in,
                              void* d_out, int out_size, void* d_ws, size_t ws_size,
                              hipStream_t stream) {
  (void)in_sizes; (void)n_in; (void)out_size; (void)ws_size;
  const float* x    = (const float*)d_in[0];
  const float* wqkv = (const float*)d_in[1];
  const float* wout = (const float*)d_in[2];
  const float* bout = (const float*)d_in[3];
  float* out = (float*)d_out;

  const size_t QKV_BYTES = (size_t)BATCH * NHEADS * HDIM * WDIM * DHEAD * sizeof(bf16);
  char* ws = (char*)d_ws;
  bf16* q  = (bf16*)(ws + 0 * QKV_BYTES);
  bf16* k  = (bf16*)(ws + 1 * QKV_BYTES);
  bf16* v  = (bf16*)(ws + 2 * QKV_BYTES);
  bf16* oh = (bf16*)(ws + 3 * QKV_BYTES);
  bf16* P  = (bf16*)(ws + 4 * QKV_BYTES);

  qkv_proj_kernel<<<dim3(NPIX / 64, 96 / 8, BATCH), 256, 0, stream>>>(x, wqkv, q, k, v);
  attn_scores_kernel<<<dim3(BATCH * NHEADS), 256, 0, stream>>>(q, k, P, 0);
  attn_pv_kernel<<<dim3(FEAT / 128, BATCH * NHEADS), 256, 0, stream>>>(P, v, oh, 0);
  attn_scores_kernel<<<dim3(BATCH * NHEADS), 256, 0, stream>>>(q, k, P, 1);
  attn_pv_kernel<<<dim3(FEAT / 128, BATCH * NHEADS), 256, 0, stream>>>(P, v, oh, 1);
  out_proj_kernel<<<dim3(NPIX / 64, 2, BATCH), 256, 0, stream>>>(oh, wout, bout, x, out);
}
